// NovaBlock_2525440770146
// MI455X (gfx1250) — matrix operand order repaired and checked by reading, not compiled, not run on hardware
//
#include <hip/hip_runtime.h>

// ---------------------------------------------------------------------------
// MI455X (gfx1250, wave32) BitLinear diff-attn + top-1 MoE block.
// Matrix ops: v_wmma_f32_16x16x32_bf16. GEMMs use a double-buffered LDS
// pipeline fed by CDNA5 async global->LDS loads (ASYNCcnt), BK=64.
// ---------------------------------------------------------------------------

#ifndef USE_ASYNC_LDS
#define USE_ASYNC_LDS 1   // 0 = VGPR-staged ping-pong fallback
#endif

typedef __attribute__((ext_vector_type(16))) __bf16        v16bf;
typedef __attribute__((ext_vector_type(8)))  float         v8f;
typedef __attribute__((ext_vector_type(4)))  unsigned int  v4u;
typedef unsigned short u16;

__device__ __forceinline__ u16 f2bf(float f) {
  unsigned int u = __float_as_uint(f);
  u += 0x7FFFu + ((u >> 16) & 1u);          // round-to-nearest-even
  return (u16)(u >> 16);
}

__device__ __forceinline__ v16bf pkbf(v4u lo, v4u hi) {
  union { v4u u[2]; v16bf v; } t;
  t.u[0] = lo; t.u[1] = hi;
  return t.v;
}

#if USE_ASYNC_LDS
// CDNA5 async DMA: per-lane 16B global -> LDS, tracked by ASYNCcnt.
// VDST = 32-bit LDS byte offset (low 32 bits of the flat shared address),
// VADDR = 64-bit global address, no SADDR ("off").
__device__ __forceinline__ void async_b128_g2l(unsigned lds_addr, const void* gp) {
  asm volatile("global_load_async_to_lds_b128 %0, %1, off"
               :: "v"(lds_addr), "v"((unsigned long long)(size_t)gp)
               : "memory");
}
__device__ __forceinline__ void wait_async0() {
  asm volatile("s_wait_asynccnt 0x0" ::: "memory");
}
#endif

// ---------------------------------------------------------------------------
// Tiled WMMA GEMM:  C[M,N] = A[M,K] * B[N,K]^T   (A,B bf16 row-major, C f32)
// Block tile 128x128, BK=64, double-buffered LDS, 8 waves (2x4),
// wave tile 64x32 -> 16 wmma per k-step, one barrier per k-step.
// EPI: 0 = f32 store, 2 = bf16(silu(acc)), 3 = scatter C[slots[off+m]]*gate,
//      4 = bf16(acc).  GA: gather A rows through slots[off+m].
// cnt_ptr (optional): effective M read on-device (fixed grid, early exit).
// Out-of-range rows are merely clamped: a stray A row only feeds the same
// C row, which the epilogue discards.
// ---------------------------------------------------------------------------
template<int EPI, bool GA>
__global__ __launch_bounds__(256) void k_gemm(
    const u16* __restrict__ A, const u16* __restrict__ Bw,
    float* __restrict__ C, u16* __restrict__ Cb,
    int M, int N, int K,
    const int* __restrict__ slots, const int* __restrict__ off_ptr,
    const int* __restrict__ cnt_ptr, const float* __restrict__ gate)
{
  __shared__ u16 As[2][128][72];   // 64 halfs + pad, stride 144B (16B aligned)
  __shared__ u16 Bs[2][128][72];
  const int mblk = blockIdx.y * 128, nblk = blockIdx.x * 128;
  int Meff = M;
  if (cnt_ptr) Meff = *cnt_ptr;
  if (mblk >= Meff) return;
  const int off = off_ptr ? *off_ptr : 0;
  const int tid = threadIdx.x;
  const int wid = tid >> 5, lane = tid & 31;
  const int wm = wid >> 2, wn = wid & 3;           // 2 x 4 waves
  const int half = lane >> 4, lr = lane & 15;

  // This thread's 4 16B slots per matrix per 64-deep k tile.
  int arow[4], btrow[4], c8v[4];
#pragma unroll
  for (int i = 0; i < 4; ++i) {
    int lin = tid + i * 256;
    int row = lin >> 3;
    c8v[i] = (lin & 7) * 8;
    int gm = mblk + row; if (gm >= Meff) gm = Meff - 1;   // clamp (row discarded later)
    arow[i] = GA ? slots[off + gm] : gm;
    btrow[i] = nblk + row;
  }

  const int nk = K >> 6;
  v4u sa[4], sb[4];
  (void)sa; (void)sb;

#if USE_ASYNC_LDS
#define TILE_ISSUE(kb, buf)                                                    \
  { _Pragma("unroll") for (int i = 0; i < 4; ++i) {                            \
      int lin = tid + i * 256; int row = lin >> 3;                             \
      async_b128_g2l((unsigned)(size_t)&As[buf][row][c8v[i]],                  \
                     A + (size_t)arow[i] * K + (kb) * 64 + c8v[i]);            \
      async_b128_g2l((unsigned)(size_t)&Bs[buf][row][c8v[i]],                  \
                     Bw + (size_t)btrow[i] * K + (kb) * 64 + c8v[i]);          \
    } }
#define TILE_COMMIT(buf) { wait_async0(); }
#else
#define TILE_ISSUE(kb, buf)                                                    \
  { _Pragma("unroll") for (int i = 0; i < 4; ++i) {                            \
      sa[i] = *(const v4u*)(A + (size_t)arow[i] * K + (kb) * 64 + c8v[i]);     \
      sb[i] = *(const v4u*)(Bw + (size_t)btrow[i] * K + (kb) * 64 + c8v[i]);   \
    } }
#define TILE_COMMIT(buf)                                                       \
  { _Pragma("unroll") for (int i = 0; i < 4; ++i) {                            \
      int lin = tid + i * 256; int row = lin >> 3;                             \
      *(v4u*)(&As[buf][row][c8v[i]]) = sa[i];                                  \
      *(v4u*)(&Bs[buf][row][c8v[i]]) = sb[i];                                  \
    } }
#endif

  v8f acc[4][2] = {};

  TILE_ISSUE(0, 0);
  TILE_COMMIT(0);
  __syncthreads();

  for (int kb = 0; kb < nk; ++kb) {
    const int cur = kb & 1;
    if (kb + 1 < nk) { TILE_ISSUE(kb + 1, cur ^ 1); }   // overlap with compute
#pragma unroll
    for (int kk = 0; kk < 2; ++kk) {                    // 2 x k32 per tile
      v16bf af[4], bfr[2];
#pragma unroll
      for (int mi = 0; mi < 4; ++mi) {                  // A 16x32 frag
        const u16* p = &As[cur][wm * 64 + mi * 16 + lr][kk * 32];
        af[mi] = pkbf(*(const v4u*)(p + 8 * half), *(const v4u*)(p + 16 + 8 * half));
      }
#pragma unroll
      for (int ni = 0; ni < 2; ++ni) {                  // B 32x16 frag
        const u16* p = &Bs[cur][wn * 32 + ni * 16 + lr][kk * 32];
        bfr[ni] = pkbf(*(const v4u*)(p + 8 * half), *(const v4u*)(p + 16 + 8 * half));
      }
#pragma unroll
      for (int mi = 0; mi < 4; ++mi)
#pragma unroll
        for (int ni = 0; ni < 2; ++ni)
          acc[mi][ni] = __builtin_amdgcn_wmma_f32_16x16x32_bf16(
              false, af[mi], false, bfr[ni], (short)0, acc[mi][ni], false, false);
    }
    if (kb + 1 < nk) { TILE_COMMIT(cur ^ 1); }
    __syncthreads();
  }
#undef TILE_ISSUE
#undef TILE_COMMIT

#pragma unroll
  for (int mi = 0; mi < 4; ++mi)
#pragma unroll
    for (int ni = 0; ni < 2; ++ni)
#pragma unroll
      for (int j = 0; j < 8; ++j) {                     // C: M=j+8*half, N=lr
        int r = mblk + wm * 64 + mi * 16 + j + 8 * half;
        int c = nblk + wn * 32 + ni * 16 + lr;
        if (r < Meff) {
          float v = acc[mi][ni][j];
          if (EPI == 0) C[(size_t)r * N + c] = v;
          else if (EPI == 2) { float s = v / (1.f + __expf(-v)); Cb[(size_t)r * N + c] = f2bf(s); }
          else if (EPI == 3) { int tok = slots[off + r]; C[(size_t)tok * N + c] = v * gate[tok]; }
          else if (EPI == 4) Cb[(size_t)r * N + c] = f2bf(v);
        }
      }
}

// ---------------------------------------------------------------------------
// Flash attention, one workgroup per (b, head, branch, 64-row q block).
// QK^T and PV on WMMA; online softmax with 4 threads per query row, staged
// through the LDS tile pads; next K/V block prefetched during compute.
// ---------------------------------------------------------------------------
__global__ __launch_bounds__(256) void k_flash(
    const u16* __restrict__ Q, const u16* __restrict__ Kb,
    const u16* __restrict__ V, float* __restrict__ O)
{
  __shared__ u16  Qs[64][72];
  __shared__ u16  Ks[64][72];
  __shared__ u16  Vt[64][72];    // V transposed: Vt[d][key]
  __shared__ u16  Ps[64][72];
  __shared__ float Sf[64][68];   // cols 64..67 = per-row max scratch
  __shared__ float Of[64][68];   // cols 64..67 = per-row sum scratch
  __shared__ float mrow[64], lrow[64];

  const int tid = threadIdx.x;
  const int qb = blockIdx.x & 15;
  const int br = (blockIdx.x >> 4) & 1;
  const int h  = (blockIdx.x >> 5) & 15;
  const int b  = blockIdx.x >> 9;

  const int lane = tid & 31, wid = tid >> 5;
  const int half = lane >> 4, lr = lane & 15;
  const int sm = wid >> 1, sn0 = (wid & 1) * 2;    // 4x4 16-tiles over 8 waves

  for (int i = tid; i < 64 * 64; i += 256) Of[i >> 6][i & 63] = 0.f;
  if (tid < 64) { mrow[tid] = -1e30f; lrow[tid] = 0.f; }
#pragma unroll
  for (int i = 0; i < 2; ++i) {
    int lin = tid + i * 256, r = lin >> 3, c8 = lin & 7;
    *(v4u*)(&Qs[r][c8 * 8]) = *(const v4u*)(
        Q + (size_t)(b * 1024 + qb * 64 + r) * 2048 + br * 1024 + h * 64 + c8 * 8);
  }
  __syncthreads();

  for (int kb = 0; kb <= qb; ++kb) {               // causal: lower blocks only
#pragma unroll
    for (int i = 0; i < 2; ++i) {
      int lin = tid + i * 256, r = lin >> 3, c8 = lin & 7;
      *(v4u*)(&Ks[r][c8 * 8]) = *(const v4u*)(
          Kb + (size_t)(b * 1024 + kb * 64 + r) * 2048 + br * 1024 + h * 64 + c8 * 8);
    }
    for (int i = tid; i < 64 * 64; i += 256) {     // transpose V into LDS
      int r = i >> 6, d = i & 63;
      Vt[d][r] = V[(size_t)(b * 1024 + kb * 64 + r) * 1024 + h * 64 + d];
    }
    if (kb < qb && tid < 64) {                     // prefetch next block rows
      __builtin_prefetch(Kb + (size_t)(b * 1024 + (kb + 1) * 64 + tid) * 2048
                            + br * 1024 + h * 64, 0, 0);
      __builtin_prefetch(V + (size_t)(b * 1024 + (kb + 1) * 64 + tid) * 1024
                           + h * 64, 0, 0);
    }
    __syncthreads();

    // S = Q K^T (scaled)
    v8f sacc[2] = {};
#pragma unroll
    for (int kk = 0; kk < 2; ++kk) {               // DH=64 -> 2 k-steps
      int qr = sm * 16 + lr;
      v16bf aq = pkbf(*(const v4u*)(&Qs[qr][kk * 32 + 8 * half]),
                      *(const v4u*)(&Qs[qr][kk * 32 + 16 + 8 * half]));
#pragma unroll
      for (int ni = 0; ni < 2; ++ni) {
        int kr = (sn0 + ni) * 16 + lr;
        v16bf bk = pkbf(*(const v4u*)(&Ks[kr][kk * 32 + 8 * half]),
                        *(const v4u*)(&Ks[kr][kk * 32 + 16 + 8 * half]));
        sacc[ni] = __builtin_amdgcn_wmma_f32_16x16x32_bf16(
            false, aq, false, bk, (short)0, sacc[ni], false, false);
      }
    }
#pragma unroll
    for (int ni = 0; ni < 2; ++ni)
#pragma unroll
      for (int j = 0; j < 8; ++j)
        Sf[sm * 16 + j + 8 * half][(sn0 + ni) * 16 + lr] = sacc[ni][j] * 0.125f;
    __syncthreads();

    // online softmax: 4 threads per query row, 16 cols each
    {
      const int r = tid >> 2, qq = tid & 3;
      const int gq = qb * 64 + r;
      int kmax = gq - kb * 64 + 1; if (kmax > 64) kmax = 64;
      float mx = -3e38f;
      for (int c = qq * 16; c < qq * 16 + 16; ++c)
        if (c < kmax) { float s = Sf[r][c]; if (s > mx) mx = s; }
      Sf[r][64 + qq] = mx;
      __syncthreads();
      const float mold = mrow[r];
      const float mnew = fmaxf(fmaxf(fmaxf(Sf[r][64], Sf[r][65]),
                                     fmaxf(Sf[r][66], Sf[r][67])), mold);
      float ls = 0.f;
      for (int c = qq * 16; c < qq * 16 + 16; ++c) {
        float p = (c < kmax) ? __expf(Sf[r][c] - mnew) : 0.f;
        Ps[r][c] = f2bf(p);
        ls += p;
      }
      Of[r][64 + qq] = ls;
      __syncthreads();
      const float alpha = __expf(mold - mnew);
      if (qq == 0) {
        lrow[r] = lrow[r] * alpha + Of[r][64] + Of[r][65] + Of[r][66] + Of[r][67];
        mrow[r] = mnew;
      }
      for (int d = qq * 16; d < qq * 16 + 16; ++d) Of[r][d] *= alpha;
    }
    __syncthreads();

    // O += P V
    v8f oacc[2] = {};
#pragma unroll
    for (int kk = 0; kk < 2; ++kk) {               // 64 keys -> 2 k-steps
      int pr = sm * 16 + lr;
      v16bf ap = pkbf(*(const v4u*)(&Ps[pr][kk * 32 + 8 * half]),
                      *(const v4u*)(&Ps[pr][kk * 32 + 16 + 8 * half]));
#pragma unroll
      for (int ni = 0; ni < 2; ++ni) {
        int vr = (sn0 + ni) * 16 + lr;
        v16bf bv = pkbf(*(const v4u*)(&Vt[vr][kk * 32 + 8 * half]),
                        *(const v4u*)(&Vt[vr][kk * 32 + 16 + 8 * half]));
        oacc[ni] = __builtin_amdgcn_wmma_f32_16x16x32_bf16(
            false, ap, false, bv, (short)0, oacc[ni], false, false);
      }
    }
#pragma unroll
    for (int ni = 0; ni < 2; ++ni)
#pragma unroll
      for (int j = 0; j < 8; ++j)
        Of[sm * 16 + j + 8 * half][(sn0 + ni) * 16 + lr] += oacc[ni][j];
    __syncthreads();
  }

  for (int i = tid; i < 64 * 64; i += 256) {
    int r = i >> 6, d = i & 63;
    O[((size_t)br * 2048 + b * 1024 + qb * 64 + r) * 1024 + h * 64 + d]
        = Of[r][d] * (1.f / lrow[r]);
  }
}

// ---------------------------------------------------------------------------
// LayerNorm (one row per block), optional f32 and/or bf16 outputs.
// ---------------------------------------------------------------------------
__global__ __launch_bounds__(256) void k_layernorm(
    const float* __restrict__ X, const float* __restrict__ g,
    const float* __restrict__ bt, float* __restrict__ Of32,
    u16* __restrict__ Obf, int D)
{
  __shared__ float red[256];
  const int tid = threadIdx.x;
  const float* xr = X + (size_t)blockIdx.x * D;
  float s = 0.f;
  for (int i = tid; i < D; i += 256) s += xr[i];
  red[tid] = s; __syncthreads();
  for (int st = 128; st > 0; st >>= 1) { if (tid < st) red[tid] += red[tid + st]; __syncthreads(); }
  float mean = red[0] / D;
  __syncthreads();
  float v = 0.f;
  for (int i = tid; i < D; i += 256) { float d = xr[i] - mean; v += d * d; }
  red[tid] = v; __syncthreads();
  for (int st = 128; st > 0; st >>= 1) { if (tid < st) red[tid] += red[tid + st]; __syncthreads(); }
  float rstd = rsqrtf(red[0] / D + 1e-5f);
  for (int i = tid; i < D; i += 256) {
    float y = (xr[i] - mean) * rstd * g[i] + bt[i];
    if (Of32) Of32[(size_t)blockIdx.x * D + i] = y;
    if (Obf)  Obf[(size_t)blockIdx.x * D + i] = f2bf(y);
  }
}

// Ternary bitlinear quantization: per-row scale = max(mean|W|, 1e-5),
// Wq = clip(rint(W/scale),-1,1)*scale  -> bf16.
__global__ __launch_bounds__(256) void k_quant(
    const float* __restrict__ W, u16* __restrict__ Wq, int cols)
{
  __shared__ float red[256];
  const int tid = threadIdx.x;
  const float* wr = W + (size_t)blockIdx.x * cols;
  float s = 0.f;
  for (int i = tid; i < cols; i += 256) s += fabsf(wr[i]);
  red[tid] = s; __syncthreads();
  for (int st = 128; st > 0; st >>= 1) { if (tid < st) red[tid] += red[tid + st]; __syncthreads(); }
  float scale = red[0] / cols;
  if (scale < 1e-5f) scale = 1e-5f;
  for (int i = tid; i < cols; i += 256) {
    float q = rintf(wr[i] / scale);
    q = fminf(1.f, fmaxf(-1.f, q));
    Wq[(size_t)blockIdx.x * cols + i] = f2bf(q * scale);
  }
}

__global__ __launch_bounds__(256) void k_lambda(
    const float* __restrict__ lq, const float* __restrict__ lk, float* lam)
{
  __shared__ float r1[256], r2[256];
  const int tid = threadIdx.x;
  float a = 0.f, b = 0.f;
  for (int i = tid; i < 1024; i += 256) { a += lq[i]; b += lk[i]; }
  r1[tid] = a; r2[tid] = b; __syncthreads();
  for (int st = 128; st > 0; st >>= 1) {
    if (tid < st) { r1[tid] += r1[tid + st]; r2[tid] += r2[tid + st]; }
    __syncthreads();
  }
  if (tid == 0) {
    float l = __expf((r1[0] - r2[0]) / 1024.f);
    lam[0] = fminf(2.f, fmaxf(0.1f, l));
  }
}

// Router: logits = h . rw[e], softmax over 8, top-1 id + prob.
__global__ __launch_bounds__(256) void k_router(
    const float* __restrict__ H, const float* __restrict__ RW,
    int* __restrict__ eidx, float* __restrict__ gate)
{
  __shared__ float part[256];
  const int tid = threadIdx.x;
  const int e = tid & 7, l = tid >> 3;              // 8 experts x 32 lanes
  const float* hr = H + (size_t)blockIdx.x * 1024;
  float s = 0.f;
  for (int i = l; i < 1024; i += 32) s += hr[i] * RW[e * 1024 + i];
  part[tid] = s; __syncthreads();
  for (int st = 16; st > 0; st >>= 1) {
    if (l < st) part[tid] += part[tid + st * 8];
    __syncthreads();
  }
  if (tid == 0) {
    float mx = part[0]; int bi = 0;
    for (int i = 1; i < 8; ++i) if (part[i] > mx) { mx = part[i]; bi = i; }
    float sum = 0.f;
    for (int i = 0; i < 8; ++i) sum += __expf(part[i] - mx);
    eidx[blockIdx.x] = bi;
    gate[blockIdx.x] = 1.f / sum;                   // top prob
  }
}

__global__ void k_zero8(int* cnt, int* cur) {
  int t = threadIdx.x; if (t < 8) { cnt[t] = 0; cur[t] = 0; }
}
__global__ void k_count(const int* eidx, int* cnt) {
  int t = blockIdx.x * 256 + threadIdx.x;
  atomicAdd(&cnt[eidx[t]], 1);
}
__global__ void k_prefix(const int* cnt, int* offs) {
  if (threadIdx.x == 0) { int s = 0; for (int e = 0; e < 8; ++e) { offs[e] = s; s += cnt[e]; } }
}
__global__ void k_compact(const int* eidx, int* cur, const int* offs, int* slots) {
  int t = blockIdx.x * 256 + threadIdx.x;
  int e = eidx[t];
  int p = atomicAdd(&cur[e], 1);
  slots[offs[e] + p] = t;                           // order-free: scatter rows unique
}
__global__ void k_combine(const float* __restrict__ o, const float* __restrict__ lamp,
                          u16* __restrict__ att) {
  int i = blockIdx.x * 256 + threadIdx.x;
  att[i] = f2bf(o[i] - lamp[0] * o[2048 * 1024 + i]);
}
__global__ void k_add2(const float* a, const float* b, float* o) {
  int i = blockIdx.x * 256 + threadIdx.x; o[i] = a[i] + b[i];
}
__global__ void k_add3(const float* a, const float* b, const float* c, float* o) {
  int i = blockIdx.x * 256 + threadIdx.x; o[i] = a[i] + b[i] + c[i];
}

// ---------------------------------------------------------------------------
extern "C" void kernel_launch(void* const* d_in, const int* in_sizes, int n_in,
                              void* d_out, int out_size, void* d_ws, size_t ws_size,
                              hipStream_t stream)
{
  (void)in_sizes; (void)n_in; (void)out_size; (void)ws_size;
  const float* x      = (const float*)d_in[0];
  // d_in[1] = causal mask (implicit in flash kernel)
  const float* Wq     = (const float*)d_in[2];
  const float* Wk     = (const float*)d_in[3];
  const float* Wv     = (const float*)d_in[4];
  const float* Wo     = (const float*)d_in[5];
  const float* lq     = (const float*)d_in[6];
  const float* lk     = (const float*)d_in[7];
  const float* qn_g   = (const float*)d_in[8];
  const float* qn_b   = (const float*)d_in[9];
  const float* kn_g   = (const float*)d_in[10];
  const float* kn_b   = (const float*)d_in[11];
  const float* attn_g = (const float*)d_in[12];
  const float* attn_b = (const float*)d_in[13];
  const float* ffn_g  = (const float*)d_in[14];
  const float* ffn_b  = (const float*)d_in[15];
  const float* moe_g  = (const float*)d_in[16];
  const float* moe_b  = (const float*)d_in[17];
  const float* sw1    = (const float*)d_in[18];
  const float* sw2    = (const float*)d_in[19];
  const float* ew1    = (const float*)d_in[20];
  const float* ew2    = (const float*)d_in[21];
  const float* rw     = (const float*)d_in[22];
  float* out = (float*)d_out;

  char* w = (char*)d_ws;
  size_t off = 0;
  auto alloc = [&](size_t bytes) -> void* {
    void* p = w + off; off += (bytes + 255) & ~(size_t)255; return p;
  };
  const size_t M = 2048;                             // B*T tokens
  u16*   wq_q   = (u16*)alloc(2048ull * 1024 * 2);
  u16*   wk_q   = (u16*)alloc(2048ull * 1024 * 2);
  u16*   wv_q   = (u16*)alloc(1024ull * 1024 * 2);
  u16*   wo_q   = (u16*)alloc(1024ull * 1024 * 2);
  u16*   sw1_q  = (u16*)alloc(2048ull * 1024 * 2);
  u16*   sw2_q  = (u16*)alloc(1024ull * 2048 * 2);
  u16*   ew1_q  = (u16*)alloc(2048ull * 1024 * 2);  // reused per expert (stream-ordered)
  u16*   ew2_q  = (u16*)alloc(1024ull * 2048 * 2);
  u16*   h_bf   = (u16*)alloc(M * 1024 * 2);
  float* gbuf   = (float*)alloc(M * 2048 * 4);      // qraw/kraw/attnproj/h2 scratch
  u16*   q_bf   = (u16*)alloc(M * 2048 * 2);
  u16*   k_bf   = (u16*)alloc(M * 2048 * 2);
  u16*   v_bf   = (u16*)alloc(M * 1024 * 2);
  float* obuf   = (float*)alloc(2 * M * 1024 * 4);  // both diff branches
  u16*   att_bf = (u16*)alloc(M * 1024 * 2);
  float* x1     = (float*)alloc(M * 1024 * 4);
  float* hm     = (float*)alloc(M * 1024 * 4);
  u16*   hm_bf  = (u16*)alloc(M * 1024 * 2);
  u16*   mid_bf = (u16*)alloc(M * 2048 * 2);
  float* shout  = (float*)alloc(M * 1024 * 4);
  float* routed = (float*)alloc(M * 1024 * 4);
  float* gate   = (float*)alloc(M * 4);
  int*   eidx   = (int*)alloc(M * 4);
  int*   slots  = (int*)alloc(M * 4);
  int*   cnt    = (int*)alloc(64);
  int*   cur    = (int*)alloc(64);
  int*   offs   = (int*)alloc(64);
  float* lam    = (float*)alloc(4);

  // ---- attention path ----
  k_layernorm<<<2048, 256, 0, stream>>>(x, attn_g, attn_b, nullptr, h_bf, 1024);
  k_quant<<<2048, 256, 0, stream>>>(Wq, wq_q, 1024);
  k_quant<<<2048, 256, 0, stream>>>(Wk, wk_q, 1024);
  k_quant<<<1024, 256, 0, stream>>>(Wv, wv_q, 1024);
  k_quant<<<1024, 256, 0, stream>>>(Wo, wo_q, 1024);
  k_quant<<<2048, 256, 0, stream>>>(sw1, sw1_q, 1024);
  k_quant<<<1024, 256, 0, stream>>>(sw2, sw2_q, 2048);

  k_gemm<0,false><<<dim3(16,16), 256, 0, stream>>>(h_bf, wq_q, gbuf, nullptr,
      2048, 2048, 1024, nullptr, nullptr, nullptr, nullptr);
  k_layernorm<<<2048, 256, 0, stream>>>(gbuf, qn_g, qn_b, nullptr, q_bf, 2048);
  k_gemm<0,false><<<dim3(16,16), 256, 0, stream>>>(h_bf, wk_q, gbuf, nullptr,
      2048, 2048, 1024, nullptr, nullptr, nullptr, nullptr);
  k_layernorm<<<2048, 256, 0, stream>>>(gbuf, kn_g, kn_b, nullptr, k_bf, 2048);
  k_gemm<4,false><<<dim3(8,16), 256, 0, stream>>>(h_bf, wv_q, nullptr, v_bf,
      2048, 1024, 1024, nullptr, nullptr, nullptr, nullptr);

  k_lambda<<<1, 256, 0, stream>>>(lq, lk, lam);
  k_flash<<<1024, 256, 0, stream>>>(q_bf, k_bf, v_bf, obuf);
  k_combine<<<8192, 256, 0, stream>>>(obuf, lam, att_bf);
  k_gemm<0,false><<<dim3(8,16), 256, 0, stream>>>(att_bf, wo_q, gbuf, nullptr,
      2048, 1024, 1024, nullptr, nullptr, nullptr, nullptr);
  k_add2<<<8192, 256, 0, stream>>>(x, gbuf, x1);

  // ---- MoE path ----
  k_layernorm<<<2048, 256, 0, stream>>>(x1, ffn_g, ffn_b, gbuf, nullptr, 1024);
  k_layernorm<<<2048, 256, 0, stream>>>(gbuf, moe_g, moe_b, hm, hm_bf, 1024);
  k_gemm<2,false><<<dim3(16,16), 256, 0, stream>>>(hm_bf, sw1_q, nullptr, mid_bf,
      2048, 2048, 1024, nullptr, nullptr, nullptr, nullptr);
  k_gemm<0,false><<<dim3(8,16), 256, 0, stream>>>(mid_bf, sw2_q, shout, nullptr,
      2048, 1024, 2048, nullptr, nullptr, nullptr, nullptr);

  k_router<<<2048, 256, 0, stream>>>(hm, rw, eidx, gate);
  k_zero8<<<1, 32, 0, stream>>>(cnt, cur);
  k_count<<<8, 256, 0, stream>>>(eidx, cnt);
  k_prefix<<<1, 32, 0, stream>>>(cnt, offs);
  k_compact<<<8, 256, 0, stream>>>(eidx, cur, offs, slots);

  for (int e = 0; e < 8; ++e) {
    k_quant<<<2048, 256, 0, stream>>>(ew1 + (size_t)e * 2048 * 1024, ew1_q, 1024);
    k_quant<<<1024, 256, 0, stream>>>(ew2 + (size_t)e * 1024 * 2048, ew2_q, 2048);
    // gathered tokens -> silu(bf16) mid (compact rows), count-guarded
    k_gemm<2,true><<<dim3(16,16), 256, 0, stream>>>(hm_bf, ew1_q, nullptr, mid_bf,
        2048, 2048, 1024, slots, offs + e, cnt + e, nullptr);
    // compact mid -> scatter back to token rows, scaled by top-1 prob
    k_gemm<3,false><<<dim3(8,16), 256, 0, stream>>>(mid_bf, ew2_q, routed, nullptr,
        2048, 1024, 2048, slots, offs + e, cnt + e, gate);
  }

  k_add3<<<8192, 256, 0, stream>>>(x1, shout, routed, out);
}
